// HyperLatticeBlock_46291157516390
// MI455X (gfx1250) — compile-verified, matrix operand order checked by reading, not run on hardware
//
#include <hip/hip_runtime.h>
#include <hip/hip_bf16.h>

typedef __bf16 bf16;
typedef __attribute__((ext_vector_type(16))) __bf16 v16bf;
typedef __attribute__((ext_vector_type(8)))  float  v8f;
typedef __attribute__((ext_vector_type(4)))  unsigned int u32x4;
typedef __attribute__((ext_vector_type(4)))  int i32x4;
typedef __attribute__((ext_vector_type(8)))  int i32x8;

struct alignas(16) U128   { unsigned int u[4]; };
struct alignas(16) U128x2 { U128 a, b; };
struct alignas(16) Bf16x8 { bf16 e[8]; };

#define D      512
#define S      48
#define DD     (S * D)         // 24576 elements per batch slab
#define NTI    32              // N tiles of 16
#define WAVES  8

// ---------------- workspace layout (bf16 elements unless noted) ----------------
#define WS_WNT   0             // wnt_t  [512][512]
#define WS_WE1   262144        // we1_t  [512][1024]
#define WS_WE2   786432        // we2_t  [512][512]
#define WS_WO    1048576       // wo_t   [512][512]
#define WS_WG    1310720       // wg_t   [512][1024]
#define WS_ADJ   1835008       // adjp   [48][64] (zero padded K=48..63)
#define WS_HNB_B 3676160       // has_nb int[48] (byte offset)

// ---------------- LDS layout (bytes) ----------------
// XFb  bf16[48][512]  @ 0        (49152)  x_front in bf16 (GEMM A operand)
// WT   bf16[48][512]  @ 49152    (49152)  weighted -> later messages
// XTT  bf16[512][48]  @ 98304    (49152)  x_t transposed (K-pad overread lands in XT: finite)
// XT   bf16[48][512]  @ 147456   (49152)  x_t
// H    bf16[48][512]  @ 196608   (49152)  hidden
// XFstage f32[48][512] @ 147456  (98304)  TDM staging, aliases XT+H (phase 0 only)
#define OFF_XFB   0
#define OFF_WT    49152
#define OFF_XTT   98304
#define OFF_XT    147456
#define OFF_H     196608
#define OFF_STAGE 147456
#define SMEM_BYTES 245760

__device__ __forceinline__ v8f wmma_bf16(v16bf a, v16bf b, v8f c) {
  return __builtin_amdgcn_wmma_f32_16x16x32_bf16(false, a, false, b, (short)0, c, false, false);
}

// A/B fragment from a bf16 row-major matrix row pointer (already offset by k0).
// 16-bit 16x32 layout: base8 = 8*(lane>=16); elements 0..7 = K base8..base8+7,
// elements 8..15 = K base8+16..base8+23. Two 16-byte loads.
__device__ __forceinline__ v16bf frag_bf16(const bf16* rowk, int kh) {
  U128x2 t;
  t.a = *(const U128*)(rowk + kh * 8);
  t.b = *(const U128*)(rowk + kh * 8 + 16);
  return __builtin_bit_cast(v16bf, t);
}

__device__ __forceinline__ float gelu_exact(float v) {
  return 0.5f * v * (1.0f + erff(v * 0.70710678118654752f));
}

// -------------------------------------------------------------------------
// Prep: convert W (K x 512, f32) -> transposed bf16 dst (512 x K)
__global__ void prep_w(const float* __restrict__ src, bf16* __restrict__ dst, int K) {
  int idx = blockIdx.x * 256 + threadIdx.x;
  if (idx >= K * 512) return;
  int k = idx >> 9, n = idx & 511;
  dst[(size_t)n * K + k] = (bf16)src[idx];
}

// Prep: softmax rows of 48x48 lattice, threshold, pad to 48x64 bf16, has_nb flags
__global__ void prep_adj(const float* __restrict__ lw, bf16* __restrict__ adjp,
                         int* __restrict__ hnb) {
  int i = threadIdx.x;
  if (i >= S) return;
  float row[S];
  float mx = -3.4e38f;
  for (int j = 0; j < S; ++j) { row[j] = lw[i * S + j]; mx = fmaxf(mx, row[j]); }
  float s = 0.f;
  for (int j = 0; j < S; ++j) { row[j] = expf(row[j] - mx); s += row[j]; }
  float inv = 1.f / s;
  int any = 0;
  for (int j = 0; j < S; ++j) {
    float a = row[j] * inv;
    int nb = a > 0.01f;
    any |= nb;
    adjp[i * 64 + j] = (bf16)(nb ? a : 0.f);
  }
  for (int j = S; j < 64; ++j) adjp[i * 64 + j] = (bf16)0.f;
  hnb[i] = any;
}

// -------------------------------------------------------------------------
__global__ __launch_bounds__(256) void hlb_main(
    const float* __restrict__ x, float* __restrict__ out,
    const bf16* __restrict__ wnt, const bf16* __restrict__ we1,
    const bf16* __restrict__ we2, const bf16* __restrict__ wo,
    const bf16* __restrict__ wg,  const bf16* __restrict__ adjp,
    const int* __restrict__ hnb,
    const float* __restrict__ b_nt, const float* __restrict__ ln_g,
    const float* __restrict__ ln_b, const float* __restrict__ b_e1,
    const float* __restrict__ b_e2, const float* __restrict__ b_o,
    const float* __restrict__ b_g)
{
  extern __shared__ char smem[];
  bf16*  XFb = (bf16*)(smem + OFF_XFB);
  bf16*  WT  = (bf16*)(smem + OFF_WT);
  bf16*  XTT = (bf16*)(smem + OFF_XTT);
  bf16*  XT  = (bf16*)(smem + OFF_XT);
  bf16*  H   = (bf16*)(smem + OFF_H);
  const float* XFstage = (const float*)(smem + OFF_STAGE);

  const int tid  = threadIdx.x;
  const int wave = tid >> 5;
  const int lane = tid & 31;
  const int kh   = lane >> 4;   // K-half selector for fragments
  const int lm   = lane & 15;   // row (A) / col (B,C)

  const float* xb = x   + (size_t)blockIdx.x * DD;
  float*       ob = out + (size_t)blockIdx.x * DD;

  // ---- Phase 0: TDM-load the 48x512 f32 tile into LDS staging, then cvt to bf16 ----
  if (wave == 0) {
    unsigned long long ga = (unsigned long long)(uintptr_t)xb;
    u32x4 g0;
    g0[0] = 1u;                                   // count=1, user descriptor
    g0[1] = (unsigned)OFF_STAGE;                  // lds_addr (bytes)
    g0[2] = (unsigned)(ga & 0xffffffffu);         // global_addr[31:0]
    g0[3] = (unsigned)((ga >> 32) & 0x01ffffffu)  // global_addr[56:32]
            | (2u << 30);                         // type=2 (image)
    i32x8 g1;
    g1[0] = 0x20000;                // wg_mask=0, data_size=2 (4 bytes)
    g1[1] = (int)(512u << 16);      // tensor_dim0[15:0] << 16  (=512)
    g1[2] = (int)(48u << 16);       // tensor_dim0 hi16=0 | tensor_dim1 lo16 (=48)
    g1[3] = (int)(512u << 16);      // tensor_dim1 hi16=0 | tile_dim0 (=512)
    g1[4] = 48;                     // tile_dim1=48, tile_dim2=0
    g1[5] = 512;                    // tensor_dim0_stride lo32 (=512)
    g1[6] = 0;                      // stride hi16 | tensor_dim1_stride lo16
    g1[7] = 0;
    i32x4 z4 = {0, 0, 0, 0};
#if defined(__clang_major__) && (__clang_major__ >= 23)
    i32x8 z8 = {0, 0, 0, 0, 0, 0, 0, 0};
    __builtin_amdgcn_tensor_load_to_lds(g0, g1, z4, z4, z8, 0);
#else
    __builtin_amdgcn_tensor_load_to_lds(g0, g1, z4, z4, 0);
#endif
    __builtin_amdgcn_s_wait_tensorcnt(0);
  }
  __syncthreads();

  // convert staging f32 -> XFb bf16 (vectorized: 2x b128 loads, 1x b128 store)
  for (int i = tid; i < DD / 8; i += 256) {
    float4 a = ((const float4*)XFstage)[2 * i];
    float4 b = ((const float4*)XFstage)[2 * i + 1];
    Bf16x8 o;
    o.e[0] = (bf16)a.x; o.e[1] = (bf16)a.y; o.e[2] = (bf16)a.z; o.e[3] = (bf16)a.w;
    o.e[4] = (bf16)b.x; o.e[5] = (bf16)b.y; o.e[6] = (bf16)b.z; o.e[7] = (bf16)b.w;
    *(Bf16x8*)(XFb + 8 * i) = o;
  }
  __syncthreads();

  // ---- Phase 1: h = x @ W_nt + b_nt -> XT ----
  for (int nt = wave; nt < NTI; nt += WAVES) {
    const int n0 = nt * 16;
    v8f a0 = {}, a1 = {}, a2 = {};
    for (int ks = 0; ks < 16; ++ks) {
      const int k0 = ks * 32;
      v16bf bfr = frag_bf16(wnt + (size_t)(n0 + lm) * D + k0, kh);
      v16bf x0  = frag_bf16(XFb + (lm)      * D + k0, kh);
      v16bf x1  = frag_bf16(XFb + (16 + lm) * D + k0, kh);
      v16bf x2  = frag_bf16(XFb + (32 + lm) * D + k0, kh);
      a0 = wmma_bf16(x0, bfr, a0);
      a1 = wmma_bf16(x1, bfr, a1);
      a2 = wmma_bf16(x2, bfr, a2);
    }
    const int n = n0 + lm;
    const float bias = b_nt[n];
#pragma unroll
    for (int r = 0; r < 8; ++r) {
      const int m = r + 8 * kh;
      XT[(m)*D + n]        = (bf16)(a0[r] + bias);
      XT[(16 + m) * D + n] = (bf16)(a1[r] + bias);
      XT[(32 + m) * D + n] = (bf16)(a2[r] + bias);
    }
  }
  __syncthreads();

  // ---- Phase 2: LayerNorm + GELU, write XT (in place) and XTT (transposed) ----
  for (int row = wave; row < S; row += WAVES) {
    float vals[16];
    float s = 0.f, s2 = 0.f;
#pragma unroll
    for (int i = 0; i < 16; ++i) {
      float v = (float)XT[row * D + lane + 32 * i];
      vals[i] = v; s += v; s2 += v * v;
    }
#pragma unroll
    for (int off = 16; off > 0; off >>= 1) {
      s  += __shfl_xor(s,  off, 32);
      s2 += __shfl_xor(s2, off, 32);
    }
    const float mu = s * (1.f / 512.f);
    const float rs = rsqrtf(s2 * (1.f / 512.f) - mu * mu + 1e-5f);
#pragma unroll
    for (int i = 0; i < 16; ++i) {
      const int c = lane + 32 * i;
      float v = (vals[i] - mu) * rs * ln_g[c] + ln_b[c];
      bf16 t = (bf16)gelu_exact(v);
      XT[row * D + c] = t;
      XTT[c * S + row] = t;
    }
  }
  __syncthreads();

  // ---- Phase 3: weighted = masked_adj @ x_t -> WT (K=48 padded to 64; adj pad is zero) ----
  for (int nt = wave; nt < NTI; nt += WAVES) {
    const int n0 = nt * 16;
    v8f a0 = {}, a1 = {}, a2 = {};
#pragma unroll
    for (int ks = 0; ks < 2; ++ks) {
      const int k0 = ks * 32;
      v16bf bfr = frag_bf16(XTT + (n0 + lm) * S + k0, kh);
      v16bf p0  = frag_bf16(adjp + (lm)      * 64 + k0, kh);
      v16bf p1  = frag_bf16(adjp + (16 + lm) * 64 + k0, kh);
      v16bf p2  = frag_bf16(adjp + (32 + lm) * 64 + k0, kh);
      a0 = wmma_bf16(p0, bfr, a0);
      a1 = wmma_bf16(p1, bfr, a1);
      a2 = wmma_bf16(p2, bfr, a2);
    }
    const int n = n0 + lm;
#pragma unroll
    for (int r = 0; r < 8; ++r) {
      const int m = r + 8 * kh;
      WT[(m)*D + n]        = (bf16)a0[r];
      WT[(16 + m) * D + n] = (bf16)a1[r];
      WT[(32 + m) * D + n] = (bf16)a2[r];
    }
  }
  __syncthreads();

  // ---- Phase 4: hidden = gelu([x_t, weighted] @ W_e1 + b_e1) -> H ----
  for (int nt = wave; nt < NTI; nt += WAVES) {
    const int n0 = nt * 16;
    v8f a0 = {}, a1 = {}, a2 = {};
    const bf16* wrow = we1 + (size_t)(n0 + lm) * 1024;
    for (int ks = 0; ks < 16; ++ks) {
      const int k0 = ks * 32;
      v16bf bfr = frag_bf16(wrow + k0, kh);
      v16bf x0  = frag_bf16(XT + (lm)      * D + k0, kh);
      v16bf x1  = frag_bf16(XT + (16 + lm) * D + k0, kh);
      v16bf x2  = frag_bf16(XT + (32 + lm) * D + k0, kh);
      a0 = wmma_bf16(x0, bfr, a0);
      a1 = wmma_bf16(x1, bfr, a1);
      a2 = wmma_bf16(x2, bfr, a2);
    }
    for (int ks = 0; ks < 16; ++ks) {
      const int k0 = ks * 32;
      v16bf bfr = frag_bf16(wrow + 512 + k0, kh);
      v16bf w0  = frag_bf16(WT + (lm)      * D + k0, kh);
      v16bf w1  = frag_bf16(WT + (16 + lm) * D + k0, kh);
      v16bf w2  = frag_bf16(WT + (32 + lm) * D + k0, kh);
      a0 = wmma_bf16(w0, bfr, a0);
      a1 = wmma_bf16(w1, bfr, a1);
      a2 = wmma_bf16(w2, bfr, a2);
    }
    const int n = n0 + lm;
    const float bias = b_e1[n];
#pragma unroll
    for (int r = 0; r < 8; ++r) {
      const int m = r + 8 * kh;
      H[(m)*D + n]        = (bf16)gelu_exact(a0[r] + bias);
      H[(16 + m) * D + n] = (bf16)gelu_exact(a1[r] + bias);
      H[(32 + m) * D + n] = (bf16)gelu_exact(a2[r] + bias);
    }
  }
  __syncthreads();

  // ---- Phase 5: msg = hidden @ W_e2 + b_e2; messages = has_nb ? msg : x_t -> WT ----
  for (int nt = wave; nt < NTI; nt += WAVES) {
    const int n0 = nt * 16;
    v8f a0 = {}, a1 = {}, a2 = {};
    for (int ks = 0; ks < 16; ++ks) {
      const int k0 = ks * 32;
      v16bf bfr = frag_bf16(we2 + (size_t)(n0 + lm) * D + k0, kh);
      v16bf h0  = frag_bf16(H + (lm)      * D + k0, kh);
      v16bf h1  = frag_bf16(H + (16 + lm) * D + k0, kh);
      v16bf h2  = frag_bf16(H + (32 + lm) * D + k0, kh);
      a0 = wmma_bf16(h0, bfr, a0);
      a1 = wmma_bf16(h1, bfr, a1);
      a2 = wmma_bf16(h2, bfr, a2);
    }
    const int n = n0 + lm;
    const float bias = b_e2[n];
#pragma unroll
    for (int r = 0; r < 8; ++r) {
      const int m = r + 8 * kh;
      WT[(m)*D + n]        = hnb[m]      ? (bf16)(a0[r] + bias) : XT[(m)*D + n];
      WT[(16 + m) * D + n] = hnb[16 + m] ? (bf16)(a1[r] + bias) : XT[(16 + m) * D + n];
      WT[(32 + m) * D + n] = hnb[32 + m] ? (bf16)(a2[r] + bias) : XT[(32 + m) * D + n];
    }
  }
  __syncthreads();

  // ---- Phase 6: g = sigmoid([x, messages] @ W_g + b_g);
  //               out = g * (messages @ W_o + b_o) + (1-g) * x ----
  for (int nt = wave; nt < NTI; nt += WAVES) {
    const int n0 = nt * 16;
    v8f g0 = {}, g1 = {}, g2 = {}, o0 = {}, o1 = {}, o2 = {};
    const bf16* wgrow = wg + (size_t)(n0 + lm) * 1024;
    const bf16* worow = wo + (size_t)(n0 + lm) * D;
    for (int ks = 0; ks < 16; ++ks) {
      const int k0 = ks * 32;
      v16bf bfr = frag_bf16(wgrow + k0, kh);
      v16bf x0  = frag_bf16(XFb + (lm)      * D + k0, kh);
      v16bf x1  = frag_bf16(XFb + (16 + lm) * D + k0, kh);
      v16bf x2  = frag_bf16(XFb + (32 + lm) * D + k0, kh);
      g0 = wmma_bf16(x0, bfr, g0);
      g1 = wmma_bf16(x1, bfr, g1);
      g2 = wmma_bf16(x2, bfr, g2);
    }
    for (int ks = 0; ks < 16; ++ks) {
      const int k0 = ks * 32;
      v16bf bg = frag_bf16(wgrow + 512 + k0, kh);
      v16bf bo = frag_bf16(worow + k0, kh);
      v16bf m0 = frag_bf16(WT + (lm)      * D + k0, kh);
      v16bf m1 = frag_bf16(WT + (16 + lm) * D + k0, kh);
      v16bf m2 = frag_bf16(WT + (32 + lm) * D + k0, kh);
      g0 = wmma_bf16(m0, bg, g0);
      g1 = wmma_bf16(m1, bg, g1);
      g2 = wmma_bf16(m2, bg, g2);
      o0 = wmma_bf16(m0, bo, o0);
      o1 = wmma_bf16(m1, bo, o1);
      o2 = wmma_bf16(m2, bo, o2);
    }
    const int n = n0 + lm;
    const float bgv = b_g[n];
    const float bov = b_o[n];
#pragma unroll
    for (int r = 0; r < 8; ++r) {
      const int m = r + 8 * kh;
      {
        float g  = 1.f / (1.f + expf(-(g0[r] + bgv)));
        float o  = o0[r] + bov;
        float xf = xb[(m)*D + n];               // residual from global (L2 hit)
        ob[(m)*D + n] = g * o + (1.f - g) * xf;
      }
      {
        float g  = 1.f / (1.f + expf(-(g1[r] + bgv)));
        float o  = o1[r] + bov;
        float xf = xb[(16 + m) * D + n];
        ob[(16 + m) * D + n] = g * o + (1.f - g) * xf;
      }
      {
        float g  = 1.f / (1.f + expf(-(g2[r] + bgv)));
        float o  = o2[r] + bov;
        float xf = xb[(32 + m) * D + n];
        ob[(32 + m) * D + n] = g * o + (1.f - g) * xf;
      }
    }
  }
}

// -------------------------------------------------------------------------
extern "C" void kernel_launch(void* const* d_in, const int* in_sizes, int n_in,
                              void* d_out, int out_size, void* d_ws, size_t ws_size,
                              hipStream_t stream) {
  (void)in_sizes; (void)n_in; (void)out_size; (void)ws_size;
  const float* x    = (const float*)d_in[0];
  const float* lw   = (const float*)d_in[1];
  const float* W_nt = (const float*)d_in[2];
  const float* b_nt = (const float*)d_in[3];
  const float* ln_g = (const float*)d_in[4];
  const float* ln_b = (const float*)d_in[5];
  const float* W_e1 = (const float*)d_in[6];
  const float* b_e1 = (const float*)d_in[7];
  const float* W_e2 = (const float*)d_in[8];
  const float* b_e2 = (const float*)d_in[9];
  const float* W_o  = (const float*)d_in[10];
  const float* b_o  = (const float*)d_in[11];
  const float* W_g  = (const float*)d_in[12];
  const float* b_g  = (const float*)d_in[13];
  float* out = (float*)d_out;

  bf16* ws   = (bf16*)d_ws;
  bf16* wnt  = ws + WS_WNT;
  bf16* we1  = ws + WS_WE1;
  bf16* we2  = ws + WS_WE2;
  bf16* wo   = ws + WS_WO;
  bf16* wg   = ws + WS_WG;
  bf16* adjp = ws + WS_ADJ;
  int*  hnb  = (int*)((char*)d_ws + WS_HNB_B);

  prep_w<<<(262144 + 255) / 256, 256, 0, stream>>>(W_nt, wnt, 512);
  prep_w<<<(524288 + 255) / 256, 256, 0, stream>>>(W_e1, we1, 1024);
  prep_w<<<(262144 + 255) / 256, 256, 0, stream>>>(W_e2, we2, 512);
  prep_w<<<(262144 + 255) / 256, 256, 0, stream>>>(W_o,  wo,  512);
  prep_w<<<(524288 + 255) / 256, 256, 0, stream>>>(W_g,  wg,  1024);
  prep_adj<<<1, 64, 0, stream>>>(lw, adjp, hnb);

  hipFuncSetAttribute(reinterpret_cast<const void*>(hlb_main),
                      hipFuncAttributeMaxDynamicSharedMemorySize, SMEM_BYTES);
  hlb_main<<<4096, 256, SMEM_BYTES, stream>>>(x, out, wnt, we1, we2, wo, wg, adjp, hnb,
                                              b_nt, ln_g, ln_b, b_e1, b_e2, b_o, b_g);
}
